// _MXFPLinearBase_64828236366439
// MI455X (gfx1250) — matrix operand order repaired and checked by reading, not compiled, not run on hardware
//
#include <hip/hip_runtime.h>
#include <hip/hip_bf16.h>

typedef __attribute__((ext_vector_type(16))) __bf16 v16bf;
typedef __attribute__((ext_vector_type(8)))  float  v8f;
typedef unsigned short u16;

#define N_ROWS 128
#define IN_F   1024
#define OUT_F  512
#define NB     32   // blocks per row
#define BS     32   // block size
#define TILE   16

union Frag {
  v16bf v;
  uint4 q[2];
};

static __device__ __forceinline__ int imin(int a, int b) { return a < b ? a : b; }
static __device__ __forceinline__ int imax(int a, int b) { return a > b ? a : b; }

// ---------------------------------------------------------------------------
// Per-block quantization: one wave32 handles one 32-element MX block.
// ---------------------------------------------------------------------------
__global__ __launch_bounds__(256)
void mxfp_quant_kernel(const float* __restrict__ src, int total_blocks,
                       u16* __restrict__ qout, float* __restrict__ scout,
                       float* __restrict__ lsout, int* __restrict__ emout)
{
  const int lane = threadIdx.x & 31;
  const int blk  = blockIdx.x * (blockDim.x >> 5) + (threadIdx.x >> 5);
  if (blk >= total_blocks) return;
  const int row = blk / NB;
  const int b   = blk - row * NB;

  const float x = src[row * IN_F + b * BS + lane];
  float m = fabsf(x);
  #pragma unroll
  for (int off = 16; off; off >>= 1) m = fmaxf(m, __shfl_xor(m, off, 32));
  const float scale = fmaxf(m, 1e-30f);

  const float xn = x / scale;
  const float an = fabsf(xn);
  float qv = 0.0f;
  if (an > 0.0f) {
    const float e    = floorf(log2f(fmaxf(an, 1e-45f)));
    const float step = exp2f(e - 3.0f);
    qv = rintf(xn / step) * step;          // RNE, <=4 significand bits
  }
  const unsigned qb = __builtin_bit_cast(unsigned, qv);
  qout[row * IN_F + b * BS + lane] = (u16)(qb >> 16);   // exact bf16 pack

  int ev = (qv != 0.0f) ? (int)((qb >> 23) & 0xFF) - 127 : -60;  // NEG_LOG2
  #pragma unroll
  for (int off = 16; off; off >>= 1) ev = imax(ev, __shfl_xor(ev, off, 32));
  if (lane == 0) {
    scout[blk] = scale;
    lsout[blk] = log2f(scale);
    emout[blk] = ev;
  }
}

// ---------------------------------------------------------------------------
// Main kernel: one wave per 16x16 output tile.
//   out = sum_b  x_s * w_s * [ WMMA_bf16 blockdot  -  NAF-truncation error ]
// ---------------------------------------------------------------------------
__global__ __launch_bounds__(32)
void mxfp_linear_kernel(const u16* __restrict__ Xq, const u16* __restrict__ Wq,
                        const float* __restrict__ Xsc, const float* __restrict__ Xls,
                        const int* __restrict__ Xem,
                        const float* __restrict__ Wsc, const float* __restrict__ Wls,
                        const float* __restrict__ bias, float* __restrict__ out)
{
  __shared__ u16 ldsX[TILE][IN_F];   // 32 KB
  __shared__ u16 ldsW[TILE][IN_F];   // 32 KB  (64 KB total)

  const int lane = threadIdx.x;
  const int n0 = blockIdx.y * TILE;
  const int o0 = blockIdx.x * TILE;

  // Stage both tiles via the CDNA5 async global->LDS path (ASYNCcnt).
  // Generic LDS pointer truncated to 32 bits == LDS byte offset (flat aperture).
  for (int i = lane; i < TILE * (IN_F / 8); i += 32) {
    const int r = i >> 7;            // IN_F/8 == 128 16-byte chunks per row
    const int c = (i & 127) * 8;
    const u16* gx = &Xq[(n0 + r) * IN_F + c];
    const u16* gw = &Wq[(o0 + r) * IN_F + c];
    unsigned lx = (unsigned)(unsigned long long)&ldsX[r][c];
    unsigned lw = (unsigned)(unsigned long long)&ldsW[r][c];
    asm volatile("global_load_async_to_lds_b128 %0, %1, off"
                 :: "v"(lx), "v"(gx) : "memory");
    asm volatile("global_load_async_to_lds_b128 %0, %1, off"
                 :: "v"(lw), "v"(gw) : "memory");
  }
  asm volatile("s_wait_asynccnt 0x0" ::: "memory");
  __syncthreads();

  const int nl = lane & 15;          // N column of D / row index in both tiles
  const int hi = lane >> 4;          // lane half selects K chunks & M offset
  const int n  = n0 + nl;

  // e_max(n,o) = max_b floor(log2 xs + log2 ws)
  float cemax[8];
  #pragma unroll
  for (int r = 0; r < 8; ++r) {
    const int o = o0 + r + 8 * hi;
    float mm = -1e30f;
    for (int b = 0; b < NB; ++b)
      mm = fmaxf(mm, floorf(Xls[n * NB + b] + Wls[o * NB + b]));
    cemax[r] = mm;
  }

  float acc[8];
  #pragma unroll
  for (int r = 0; r < 8; ++r) acc[r] = bias[o0 + r + 8 * hi];

  for (int b = 0; b < NB; ++b) {
    // A = W tile (16x32). 16-bit A layout: lanes 0-15 K{0-7,16-23},
    // lanes 16-31 K{8-15,24-31}; row M = lane&15.
    Frag afrag, bfrag;
    const int ka = b * BS + hi * 8;
    afrag.q[0] = *(const uint4*)&ldsW[nl][ka];
    afrag.q[1] = *(const uint4*)&ldsW[nl][ka + 16];
    // B = X tile as 32x16: lanes 0-15 K{0-7,16-23}, lanes 16-31 K{8-15,24-31}, N = lane&15.
    const int kb = b * BS + hi * 8;
    bfrag.q[0] = *(const uint4*)&ldsX[nl][kb];
    bfrag.q[1] = *(const uint4*)&ldsX[nl][kb + 16];

    v8f c = {0.f, 0.f, 0.f, 0.f, 0.f, 0.f, 0.f, 0.f};
    c = __builtin_amdgcn_wmma_f32_16x16x32_bf16(false, afrag.v, false, bfrag.v,
                                                (short)0, c, false, false);

    const float sx  = Xsc[n * NB + b];
    const float lxb = Xls[n * NB + b];
    const int   emb = Xem[n * NB + b];

    // Per-row digit budgets and LDS row pointers (constant-indexed arrays).
    int effb[8];
    const u16* wrowp[8];
    #pragma unroll
    for (int r = 0; r < 8; ++r) {
      const int o  = o0 + r + 8 * hi;
      const int ol = r + 8 * hi;
      const int inter = (int)(cemax[r] - floorf(lxb + Wls[o * NB + b]));
      effb[r]  = 14 - inter - emb;           // eff = effb + Ex
      wrowp[r] = &ldsW[ol][b * BS];
    }
    const u16* xrowp = &ldsX[nl][b * BS];

    float errv[8];
    #pragma unroll
    for (int r = 0; r < 8; ++r) errv[r] = 0.0f;

    // Rolled chunk loop (8 x 4 elements); r unrolled inside so all arrays
    // are constant-indexed registers. Fully branchless integer NAF.
    for (int kw = 0; kw < 8; ++kw) {
      const uint2 xw = *(const uint2*)(xrowp + kw * 4);      // ds_load_b64
      uint2 ww[8];
      #pragma unroll
      for (int r = 0; r < 8; ++r)
        ww[r] = *(const uint2*)(wrowp[r] + kw * 4);          // ds_load_b64

      #pragma unroll
      for (int e = 0; e < 4; ++e) {
        const unsigned xd = (e < 2) ? xw.x : xw.y;
        const int sh = (e & 1) * 16;
        const unsigned xb = (xd >> sh) & 0xffffu;
        // x-side decode, shared by all 8 rows
        const unsigned xmag = xb & 0x7fffu;
        const int Ex  = (int)((xb >> 7) & 0xFF) - 127;
        const int mx  = 8 | ((xb >> 4) & 7);

        #pragma unroll
        for (int r = 0; r < 8; ++r) {
          const unsigned wd = (e < 2) ? ww[r].x : ww[r].y;
          const unsigned wb = (wd >> sh) & 0xffffu;
          const unsigned tt = xmag * (wb & 0x7fffu);  // ==0 iff either is zero
          const int m2   = (int)(tt | (0u - tt)) >> 31;      // -1 / 0 mask
          const int Ew   = (int)((wb >> 7) & 0xFF) - 127;
          const int mw   = 8 | ((wb >> 4) & 7);
          const int pint = (mx * mw) & m2;           // product significand
          const int eff  = effb[r] + Ex;             // digit budget (any sign)
          const int h    = pint >> 1;
          const int s    = pint + h;
          const int pos  = s & ~h;                   // NAF +1 digits
          const int neg  = h & ~s;                   // NAF -1 digits
          const int comb = (pos | neg) | 1;          // clz-safe
          const int width = 32 - __builtin_clz(comb);
          // eff<=0 => drop>=width => recon==0 (matches masked-out reference)
          const int drop = imin(imax(width - eff, 0), 31);
          const int keep = ~((1 << drop) - 1);
          const int recon  = (pos & keep) - (neg & keep);
          const int errint = pint - recon;           // 0 for most elements
          const float ef = ldexpf((float)errint, Ex + Ew - 6); // v_ldexp_f32
          const unsigned sgn = ((xb ^ wb) & 0x8000u) << 16;    // sign-xor
          errv[r] += __builtin_bit_cast(float,
                       __builtin_bit_cast(unsigned, ef) ^ sgn);
        }
      }
    }

    #pragma unroll
    for (int r = 0; r < 8; ++r) {
      const int o = o0 + r + 8 * hi;
      const float s2 = sx * Wsc[o * NB + b];
      acc[r] += s2 * (c[r] - errv[r]);
    }
  }

  #pragma unroll
  for (int r = 0; r < 8; ++r)
    out[n * OUT_F + o0 + r + 8 * hi] = acc[r];
}

// ---------------------------------------------------------------------------
extern "C" void kernel_launch(void* const* d_in, const int* in_sizes, int n_in,
                              void* d_out, int out_size, void* d_ws, size_t ws_size,
                              hipStream_t stream) {
  const float* x    = (const float*)d_in[0];
  const float* w    = (const float*)d_in[1];
  const float* bias = (const float*)d_in[2];
  float* out = (float*)d_out;

  char* p = (char*)d_ws;
  u16*   Xq  = (u16*)p;   p += (size_t)N_ROWS * IN_F * sizeof(u16);
  u16*   Wq  = (u16*)p;   p += (size_t)OUT_F  * IN_F * sizeof(u16);
  float* Xsc = (float*)p; p += (size_t)N_ROWS * NB * 4;
  float* Xls = (float*)p; p += (size_t)N_ROWS * NB * 4;
  int*   Xem = (int*)p;   p += (size_t)N_ROWS * NB * 4;
  float* Wsc = (float*)p; p += (size_t)OUT_F * NB * 4;
  float* Wls = (float*)p; p += (size_t)OUT_F * NB * 4;
  int*   Wem = (int*)p;   p += (size_t)OUT_F * NB * 4;

  const int xblocks = N_ROWS * NB;   // 4096 MX blocks
  const int wblocks = OUT_F * NB;    // 16384 MX blocks
  mxfp_quant_kernel<<<dim3((xblocks + 7) / 8), 256, 0, stream>>>(
      x, xblocks, Xq, Xsc, Xls, Xem);
  mxfp_quant_kernel<<<dim3((wblocks + 7) / 8), 256, 0, stream>>>(
      w, wblocks, Wq, Wsc, Wls, Wem);
  mxfp_linear_kernel<<<dim3(OUT_F / TILE, N_ROWS / TILE), 32, 0, stream>>>(
      Xq, Wq, Xsc, Xls, Xem, Wsc, Wls, bias, out);
}
